// GNN_44985487458615
// MI455X (gfx1250) — matrix operand order repaired and checked by reading, not compiled, not run on hardware
//
#include <hip/hip_runtime.h>
#include <stdint.h>

// ---------------- problem constants (match reference) ----------------
#define NN     100000      // nodes
#define NE     1600000     // edges
#define NB     256         // graphs
#define DIN    64
#define DEMB   128
#define GI     192         // DIN + DEMB
#define H1D    32
#define H2D    8
#define LAYERS 3
#define EPS_LN 1e-5f

typedef __attribute__((ext_vector_type(16))) __bf16 v16bf;
typedef __attribute__((ext_vector_type(8)))  __bf16 v8bf;
typedef __attribute__((ext_vector_type(8)))  float  v8f;

// ---------------- WMMA fragment helpers (wave32, 16x16x32 bf16) ------
// A (16x32, row-major, 16-bit): lanes 0-15 hold row M=lane, K=0..7 (v0-3) and
// K=16..23 (v4-7); lanes 16-31 hold K=8..15 and K=24..31.
__device__ __forceinline__ v16bf load_a_frag(const __bf16* __restrict__ A, int ld,
                                             int row, int k0, int lane) {
  const int kb = k0 + ((lane & 16) ? 8 : 0);
  const __bf16* p = A + (size_t)row * ld + kb;
  v8bf lo = *(const v8bf*)(p);        // K = kb .. kb+7
  v8bf hi = *(const v8bf*)(p + 16);   // K = kb+16 .. kb+23
  v16bf a;
#pragma unroll
  for (int j = 0; j < 8; ++j) { a[j] = lo[j]; a[8 + j] = hi[j]; }
  return a;
}

// B fragments are kept in LDS pre-swizzled: layout [kt][t][lane][16 bf16],
// so a fragment load is one contiguous 32B LDS read (2x ds_load_b128).
__device__ __forceinline__ v16bf load_b_sw(const __bf16* __restrict__ Wlds,
                                           int kt, int t, int lane) {
  return *(const v16bf*)(Wlds + ((((kt << 3) + t) << 5) + lane) * 16);
}

// Fill swizzled weight LDS from row-major fp32 W[K x 128].
// Inverse map: element (k,c) -> kt=k/32, lane=(c&15)+((k&8)?16:0), j=(k&7)+((k&16)?8:0).
__device__ __forceinline__ void fill_w_swizzled(__bf16* __restrict__ Wlds,
                                                const float* __restrict__ W,
                                                int K, int tid, int nthreads) {
  for (int idx = tid; idx < K * DEMB; idx += nthreads) {
    const int k = idx >> 7;         // idx = k*128 + c
    const int c = idx & 127;
    const int kt = k >> 5;
    const int kr = k & 31;
    const int t  = c >> 4;
    const int ln = (c & 15) + ((kr & 8) ? 16 : 0);
    const int j  = (kr & 7) + ((kr & 16) ? 8 : 0);
    Wlds[(((((kt << 3) + t) << 5) + ln) << 4) + j] = (__bf16)W[idx];
  }
}

// ---------------- elementwise / memory kernels -----------------------
__global__ void k_zero_f32(float* __restrict__ p, long long n) {
  long long i = (long long)blockIdx.x * blockDim.x + threadIdx.x;
  if (i < n) p[i] = 0.0f;
}

__global__ void k_f32_to_bf16(const float* __restrict__ s, __bf16* __restrict__ d,
                              long long n) {
  long long i = (long long)blockIdx.x * blockDim.x + threadIdx.x;
  if (i < n) d[i] = (__bf16)s[i];
}

// agg[dst] += [x,h][src] over all edges; float4 gather + f32 atomics.
__global__ void k_scatter(const float* __restrict__ x, const float* __restrict__ h,
                          const int* __restrict__ ei, float* __restrict__ agg) {
  long long idx = (long long)blockIdx.x * blockDim.x + threadIdx.x;
  if (idx >= (long long)NE * (GI / 4)) return;
  const int e = (int)(idx / (GI / 4));
  const int f = (int)(idx % (GI / 4)) * 4;
  const int src = ei[e];
  const int dst = ei[NE + e];
  float4 v;
  if (f < DIN) v = *(const float4*)(x + (size_t)src * DIN + f);
  else         v = *(const float4*)(h + (size_t)src * DEMB + (f - DIN));
  float* o = agg + (size_t)dst * GI + f;
  atomicAdd(o + 0, v.x); atomicAdd(o + 1, v.y);
  atomicAdd(o + 2, v.z); atomicAdd(o + 3, v.w);
}

// z = (1+eps_l)*[x,h] + agg, emitted as bf16 GEMM operand.
__global__ void k_build_z(const float* __restrict__ x, const float* __restrict__ h,
                          const float* __restrict__ agg, const float* __restrict__ eps,
                          int l, __bf16* __restrict__ z) {
  long long idx = (long long)blockIdx.x * blockDim.x + threadIdx.x;
  if (idx >= (long long)NN * GI) return;
  const int n = (int)(idx / GI);
  const int f = (int)(idx % GI);
  const float gin = (f < DIN) ? x[(size_t)n * DIN + f] : h[(size_t)n * DEMB + (f - DIN)];
  z[idx] = (__bf16)((1.0f + eps[l]) * gin + agg[idx]);
}

// ---------------- WMMA GEMM: [n_rows x K] @ [K x 128] + bias, ReLU ---
__global__ void __launch_bounds__(256)
k_gemm_relu(const __bf16* __restrict__ A, int K,
            const float* __restrict__ W, const float* __restrict__ bias,
            float* __restrict__ outF, __bf16* __restrict__ outB, int n_rows) {
  __shared__ __bf16 Wlds[(GI / 32) * 8 * 32 * 16];   // 6 ktiles * 8 ctiles: 48KB
  fill_w_swizzled(Wlds, W, K, threadIdx.x, 256);
  __syncthreads();

  const int lane = threadIdx.x & 31;
  const int wave = threadIdx.x >> 5;
  const int rowbase = blockIdx.x * 128 + wave * 16;
  if (rowbase >= n_rows) return;

  int arow = rowbase + (lane & 15);
  if (arow >= n_rows) arow = n_rows - 1;

  v8f acc[8];
#pragma unroll
  for (int t = 0; t < 8; ++t)
#pragma unroll
    for (int i = 0; i < 8; ++i) acc[t][i] = 0.0f;

  for (int k0 = 0; k0 < K; k0 += 32) {
    if (k0 + 32 < K)   // prefetch next A fragment (global_prefetch_b8 path)
      __builtin_prefetch(A + (size_t)arow * K + k0 + 32, 0, 1);
    const v16bf a = load_a_frag(A, K, arow, k0, lane);
    const int kt = k0 >> 5;
    // hoist all 8 B fragments: DS loads issue as a batch so the 8-WMMA
    // chain is not serialized behind per-tile s_wait_dscnt 0
    v16bf bfr[8];
#pragma unroll
    for (int t = 0; t < 8; ++t) bfr[t] = load_b_sw(Wlds, kt, t, lane);
#pragma unroll
    for (int t = 0; t < 8; ++t)
      acc[t] = __builtin_amdgcn_wmma_f32_16x16x32_bf16(
          false, a, false, bfr[t], (short)0, acc[t], false, false);
  }

  const int mofs = (lane & 16) ? 8 : 0;
#pragma unroll
  for (int t = 0; t < 8; ++t) {
    const int c = t * 16 + (lane & 15);
    const float bv = bias[c];
#pragma unroll
    for (int i = 0; i < 8; ++i) {
      const int m = rowbase + mofs + i;
      if (m < n_rows) {
        float v = acc[t][i] + bv;
        v = v > 0.0f ? v : 0.0f;
        if (outF) outF[(size_t)m * DEMB + c] = v;
        if (outB) outB[(size_t)m * DEMB + c] = (__bf16)v;
      }
    }
  }
}

// ------- WMMA GEMM (K=128) + bias + optional ReLU + LayerNorm + optional ELU
__global__ void __launch_bounds__(128)
k_gemm_ln(const __bf16* __restrict__ A,
          const float* __restrict__ W, const float* __restrict__ bias,
          const float* __restrict__ g, const float* __restrict__ bvec,
          int do_relu, int do_elu,
          float* __restrict__ outF, __bf16* __restrict__ outB, int n_rows) {
  __shared__ __bf16 Wlds[(DEMB / 32) * 8 * 32 * 16]; // 4 ktiles: 32KB
  __shared__ float  stage[4][16 * DEMB];             // 32KB row staging for LN
  fill_w_swizzled(Wlds, W, DEMB, threadIdx.x, 128);
  __syncthreads();

  const int lane = threadIdx.x & 31;
  const int wave = threadIdx.x >> 5;
  const int rowbase = blockIdx.x * 64 + wave * 16;

  v8f acc[8];
#pragma unroll
  for (int t = 0; t < 8; ++t)
#pragma unroll
    for (int i = 0; i < 8; ++i) acc[t][i] = 0.0f;

  int arow = rowbase + (lane & 15);
  if (arow >= n_rows) arow = n_rows - 1;
  if (arow < 0) arow = 0;

#pragma unroll
  for (int k0 = 0; k0 < DEMB; k0 += 32) {
    const v16bf a = load_a_frag(A, DEMB, arow, k0, lane);
    const int kt = k0 >> 5;
    v16bf bfr[8];
#pragma unroll
    for (int t = 0; t < 8; ++t) bfr[t] = load_b_sw(Wlds, kt, t, lane);
#pragma unroll
    for (int t = 0; t < 8; ++t)
      acc[t] = __builtin_amdgcn_wmma_f32_16x16x32_bf16(
          false, a, false, bfr[t], (short)0, acc[t], false, false);
  }

  const int mofs = (lane & 16) ? 8 : 0;
#pragma unroll
  for (int t = 0; t < 8; ++t) {
    const int c = t * 16 + (lane & 15);
    const float bv = bias[c];
#pragma unroll
    for (int i = 0; i < 8; ++i) {
      float v = acc[t][i] + bv;
      if (do_relu && v < 0.0f) v = 0.0f;
      stage[wave][(mofs + i) * DEMB + c] = v;
    }
  }
  __syncthreads();

  // LayerNorm: all 32 lanes active; lanes l and l^16 share row (l&15),
  // each covers 64 columns; partial sums combined with one shfl_xor.
  {
    const int row   = lane & 15;
    const int m     = rowbase + row;
    const int chalf = (lane & 16) ? 64 : 0;
    const float* r  = &stage[wave][row * DEMB];
    float s = 0.0f, s2 = 0.0f;
    for (int c = chalf; c < chalf + 64; ++c) {
      const float v = r[c];
      s += v; s2 += v * v;
    }
    s  += __shfl_xor(s, 16);
    s2 += __shfl_xor(s2, 16);
    const float mu  = s * (1.0f / DEMB);
    const float var = s2 * (1.0f / DEMB) - mu * mu;
    const float rs  = rsqrtf(var + EPS_LN);
    if (m < n_rows) {
      for (int c = chalf; c < chalf + 64; ++c) {
        float v = (r[c] - mu) * rs * g[c] + bvec[c];
        if (do_elu && v < 0.0f) v = __expf(v) - 1.0f;
        outF[(size_t)m * DEMB + c] = v;
        if (outB) outB[(size_t)m * DEMB + c] = (__bf16)v;
      }
    }
  }
}

// ---------------- pooling + small MLP heads --------------------------
__global__ void k_pool(const float* __restrict__ ge, const int* __restrict__ bidx,
                       float* __restrict__ sums, float* __restrict__ cnt) {
  long long idx = (long long)blockIdx.x * blockDim.x + threadIdx.x;
  if (idx >= (long long)NN * (DEMB / 4)) return;
  const int n = (int)(idx >> 5);
  const int f = (int)(idx & 31) * 4;
  const int b = bidx[n];
  const float4 v = *(const float4*)(ge + (size_t)n * DEMB + f);
  float* o = sums + (size_t)b * DEMB + f;
  atomicAdd(o + 0, v.x); atomicAdd(o + 1, v.y);
  atomicAdd(o + 2, v.z); atomicAdd(o + 3, v.w);
  if (f == 0) atomicAdd(cnt + b, 1.0f);
}

// nodewise decoder: [x, node_emb] (192) -> 32 -> 8 -> 1
__global__ void __launch_bounds__(256)
k_decoder(const float* __restrict__ x, const float* __restrict__ emb,
          const float* __restrict__ dW1, const float* __restrict__ db1,
          const float* __restrict__ dW2, const float* __restrict__ db2,
          const float* __restrict__ dW3, const float* __restrict__ db3,
          float* __restrict__ out) {
  __shared__ float W1[GI * H1D];   // 24KB
  __shared__ float W2[H1D * H2D];
  __shared__ float W3[H2D];
  for (int i = threadIdx.x; i < GI * H1D; i += 256) W1[i] = dW1[i];
  for (int i = threadIdx.x; i < H1D * H2D; i += 256) W2[i] = dW2[i];
  if (threadIdx.x < H2D) W3[threadIdx.x] = dW3[threadIdx.x];
  __syncthreads();

  const int n = blockIdx.x * 256 + threadIdx.x;
  if (n >= NN) return;

  float a1[H1D];
#pragma unroll
  for (int j = 0; j < H1D; ++j) a1[j] = db1[j];
  for (int k = 0; k < GI; ++k) {
    const float v = (k < DIN) ? x[(size_t)n * DIN + k]
                              : emb[(size_t)n * DEMB + (k - DIN)];
    const float* w = &W1[k * H1D];
#pragma unroll
    for (int j = 0; j < H1D; ++j) a1[j] += v * w[j];
  }
  float a2[H2D];
#pragma unroll
  for (int j = 0; j < H2D; ++j) a2[j] = db2[j];
#pragma unroll
  for (int j = 0; j < H1D; ++j) {
    const float v = a1[j] > 0.0f ? a1[j] : 0.0f;
#pragma unroll
    for (int k = 0; k < H2D; ++k) a2[k] += v * W2[j * H2D + k];
  }
  float r = db3[0];
#pragma unroll
  for (int k = 0; k < H2D; ++k) {
    const float v = a2[k] > 0.0f ? a2[k] : 0.0f;
    r += v * W3[k];
  }
  out[n] = r;
}

// graphwise head: mean-pool emb (128) -> 32 -> 8 -> 1
__global__ void k_graph_head(const float* __restrict__ sums, const float* __restrict__ cnt,
                             const float* __restrict__ gW1, const float* __restrict__ gb1,
                             const float* __restrict__ gW2, const float* __restrict__ gb2,
                             const float* __restrict__ gW3, const float* __restrict__ gb3,
                             float* __restrict__ out) {
  const int b = blockIdx.x * blockDim.x + threadIdx.x;
  if (b >= NB) return;
  const float inv = 1.0f / fmaxf(cnt[b], 1.0f);
  float a1[H1D];
#pragma unroll
  for (int j = 0; j < H1D; ++j) a1[j] = gb1[j];
  for (int k = 0; k < DEMB; ++k) {
    const float e = sums[(size_t)b * DEMB + k] * inv;
    const float* w = &gW1[k * H1D];
#pragma unroll
    for (int j = 0; j < H1D; ++j) a1[j] += e * w[j];
  }
  float a2[H2D];
#pragma unroll
  for (int j = 0; j < H2D; ++j) a2[j] = gb2[j];
#pragma unroll
  for (int j = 0; j < H1D; ++j) {
    const float v = a1[j] > 0.0f ? a1[j] : 0.0f;
#pragma unroll
    for (int k = 0; k < H2D; ++k) a2[k] += v * gW2[j * H2D + k];
  }
  float r = gb3[0];
#pragma unroll
  for (int k = 0; k < H2D; ++k) {
    const float v = a2[k] > 0.0f ? a2[k] : 0.0f;
    r += v * gW3[k];
  }
  out[b] = r;
}

// ---------------- host orchestration ---------------------------------
extern "C" void kernel_launch(void* const* d_in, const int* in_sizes, int n_in,
                              void* d_out, int out_size, void* d_ws, size_t ws_size,
                              hipStream_t stream) {
  const float* x       = (const float*)d_in[0];
  const int*   ei      = (const int*)  d_in[1];
  const int*   bidx    = (const int*)  d_in[2];
  const float* W_in    = (const float*)d_in[3];
  const float* b_in    = (const float*)d_in[4];
  const float* eps     = (const float*)d_in[5];
  const float* mlp_W1  = (const float*)d_in[6];
  const float* mlp_b1  = (const float*)d_in[7];
  const float* mlp_W2  = (const float*)d_in[8];
  const float* mlp_b2  = (const float*)d_in[9];
  const float* ln_g    = (const float*)d_in[10];
  const float* ln_b    = (const float*)d_in[11];
  const float* out_W   = (const float*)d_in[12];
  const float* out_b   = (const float*)d_in[13];
  const float* lnf_g   = (const float*)d_in[14];
  const float* lnf_b   = (const float*)d_in[15];
  const float* graph_W = (const float*)d_in[16];
  const float* graph_b = (const float*)d_in[17];
  const float* gln_g   = (const float*)d_in[18];
  const float* gln_b   = (const float*)d_in[19];
  const float* dW1 = (const float*)d_in[20]; const float* db1 = (const float*)d_in[21];
  const float* dW2 = (const float*)d_in[22]; const float* db2 = (const float*)d_in[23];
  const float* dW3 = (const float*)d_in[24]; const float* db3 = (const float*)d_in[25];
  const float* gW1 = (const float*)d_in[26]; const float* gb1 = (const float*)d_in[27];
  const float* gW2 = (const float*)d_in[28]; const float* gb2 = (const float*)d_in[29];
  const float* gW3 = (const float*)d_in[30]; const float* gb3 = (const float*)d_in[31];
  float* out = (float*)d_out;

  // workspace carve-up (256B aligned slabs)
  uint8_t* ws = (uint8_t*)d_ws;
  size_t off = 0;
  auto take = [&](size_t bytes) -> void* {
    void* p = ws + off;
    off += (bytes + 255) & ~(size_t)255;
    return p;
  };
  float*  hF   = (float*) take((size_t)NN * DEMB * 4);
  __bf16* hB   = (__bf16*)take((size_t)NN * DEMB * 2);
  __bf16* xB   = (__bf16*)take((size_t)NN * DIN * 2);
  float*  agg  = (float*) take((size_t)NN * GI * 4);
  __bf16* zB   = (__bf16*)take((size_t)NN * GI * 2);
  __bf16* tB   = (__bf16*)take((size_t)NN * DEMB * 2);
  float*  sums = (float*) take((size_t)NB * DEMB * 4);
  float*  cnt  = (float*) take((size_t)NB * 4);
  // after last GIN layer, agg / zB+tB regions are dead -> reuse:
  float* node_emb = agg;          // 51.2MB fits in 76.8MB agg slab
  float* ge       = (float*)zB;   // 51.2MB fits in zB+tB (64MB contiguous)

  auto blks = [](long long n, int b) { return (unsigned)((n + b - 1) / b); };

  // x -> bf16 once; all GEMM A-operands are bf16
  k_f32_to_bf16<<<blks((long long)NN * DIN, 256), 256, 0, stream>>>(x, xB, (long long)NN * DIN);

  // encoder: h = relu(x @ W_in + b_in)
  k_gemm_relu<<<blks(NN, 128), 256, 0, stream>>>(xB, DIN, W_in, b_in, hF, nullptr, NN);

  for (int l = 0; l < LAYERS; ++l) {
    k_zero_f32<<<blks((long long)NN * GI, 256), 256, 0, stream>>>(agg, (long long)NN * GI);
    k_scatter<<<blks((long long)NE * (GI / 4), 256), 256, 0, stream>>>(x, hF, ei, agg);
    k_build_z<<<blks((long long)NN * GI, 256), 256, 0, stream>>>(x, hF, agg, eps, l, zB);
    // t = relu(z @ W1[l] + b1[l])   (192 -> 128)
    k_gemm_relu<<<blks(NN, 128), 256, 0, stream>>>(
        zB, GI, mlp_W1 + (size_t)l * GI * DEMB, mlp_b1 + (size_t)l * DEMB,
        nullptr, tB, NN);
    // h = elu(LN(t @ W2[l] + b2[l]))  (128 -> 128); bf16 copy on last layer
    k_gemm_ln<<<blks(NN, 64), 128, 0, stream>>>(
        tB, mlp_W2 + (size_t)l * DEMB * DEMB, mlp_b2 + (size_t)l * DEMB,
        ln_g + (size_t)l * DEMB, ln_b + (size_t)l * DEMB,
        /*relu*/0, /*elu*/1, hF, (l == LAYERS - 1) ? hB : nullptr, NN);
  }

  // graph embedding path: ge = LN(relu(h @ graph_W + graph_b)) then mean-pool
  k_gemm_ln<<<blks(NN, 64), 128, 0, stream>>>(
      hB, graph_W, graph_b, gln_g, gln_b, /*relu*/1, /*elu*/0, ge, nullptr, NN);
  k_zero_f32<<<blks((long long)NB * DEMB + NB, 256), 256, 0, stream>>>(sums, (long long)NB * DEMB + NB);
  k_pool<<<blks((long long)NN * (DEMB / 4), 256), 256, 0, stream>>>(ge, bidx, sums, cnt);

  // node embedding: node_emb = LN(relu(h @ out_W + out_b))
  k_gemm_ln<<<blks(NN, 64), 128, 0, stream>>>(
      hB, out_W, out_b, lnf_g, lnf_b, /*relu*/1, /*elu*/0, node_emb, nullptr, NN);

  // decoder heads
  k_decoder<<<blks(NN, 256), 256, 0, stream>>>(x, node_emb, dW1, db1, dW2, db2, dW3, db3, out);
  k_graph_head<<<1, NB, 0, stream>>>(sums, cnt, gW1, gb1, gW2, gb2, gW3, gb3, out + NN);
}
